// NodeBlock_12017318494540
// MI455X (gfx1250) — matrix operand order repaired and checked by reading, not compiled, not run on hardware
//
#include <hip/hip_runtime.h>
#include <hip/hip_bf16.h>

// ---------------------------------------------------------------------------
// CDNA5 (gfx1250) fused MolDiff NodeBlock.
// All GEMMs via v_wmma_f32_16x16x32_f16 (wave32, 16x16 tiles, K-step 32).
// A-fragments hoisted & reused across the 8 output column tiles.
// Weights pre-permuted to fragment order -> B fragment = 2x ds_load_b128.
// ---------------------------------------------------------------------------

typedef __attribute__((ext_vector_type(16))) _Float16 v16h;
typedef __attribute__((ext_vector_type(8)))  float    v8f;

union frag16 { v16h v; unsigned u[8]; uint4 q[2]; };
union accf   { v8f  v; float    f[8]; };

#define WAVES        8
#define BLK          256
#define TILE_M       16
#define BLK_ROWS     (WAVES * TILE_M)       // 128 rows per block
#define WBUF_U32     12288                  // max packed weight (gate W1: NS=6)
#define WAVE_U32     2608                   // Sedge(512)+Sx(1024)+Sh(1024)+meta(48)
#define SMEM_U32     (WBUF_U32 + WAVES * WAVE_U32)
#define SMEM_BYTES   (SMEM_U32 * 4)

// d_in flattened order (setup_inputs dict order, nested dicts in insertion order)
enum {
  IN_X = 0, IN_EIDX, IN_EATTR, IN_NTIME,
  NN_W1, NN_B1, NN_G1, NN_BE1, NN_W2, NN_B2,
  EN_W1, EN_B1, EN_G1, EN_BE1, EN_W2, EN_B2,
  GN_W1, GN_B1, GN_G1, GN_BE1, GN_W2, GN_B2,
  P_MSGW, P_MSGB, P_CENW, P_CENB, P_LNG, P_LNB, P_OUTW, P_OUTB
};

__device__ __forceinline__ unsigned pack2(float a, float b) {
  union { _Float16 h[2]; unsigned u; } p;
  p.h[0] = (_Float16)a; p.h[1] = (_Float16)b;
  return p.u;
}
__device__ __forceinline__ float unpack_half(unsigned u, int hi) {
  union { unsigned u; _Float16 h[2]; } p; p.u = u;
  return (float)p.h[hi];
}

// A fragment: 16x32 f16 tile. S = packed pairs row-major [m][k2].
// Lane's 8 dwords are two 16B-aligned quads -> 2x ds_load_b128.
__device__ __forceinline__ v16h load_a(const unsigned* S, int strideU32, int s) {
  int l = threadIdx.x & 31, m = l & 15, half = l >> 4;
  const unsigned* r = S + m * strideU32 + s * 16 + half * 8;
  frag16 f;
  f.q[0] = *(const uint4*)r;        // K = 32s + 16*half + {0,2,4,6}(+1)
  f.q[1] = *(const uint4*)(r + 4);  // K = 32s + 16*half + 8 + {0,2,4,6}(+1)
  return f.v;
}

// B fragment from fragment-ordered weight buffer:
// Wbuf[((ct*NS + s)*32 + lane)*8 + v] = pair(K = 32s + 16*half + 2v, n)
__device__ __forceinline__ v16h load_b(const unsigned* Wbuf, int ctNSs) {
  int l = threadIdx.x & 31;
  const unsigned* r = Wbuf + ((ctNSs) * 32 + l) * 8;
  frag16 f;
  f.q[0] = *(const uint4*)r;
  f.q[1] = *(const uint4*)(r + 4);
  return f.v;
}

// GEMM k-steps with pre-loaded A fragments (reused across column tiles)
__device__ __forceinline__ void gemm_pre(accf& a, const v16h* af,
                                         const unsigned* Wbuf, int ctNS,
                                         int ws0, int ns) {
#pragma unroll
  for (int s = 0; s < ns; ++s) {
    v16h bf = load_b(Wbuf, ctNS + ws0 + s);
    a.v = __builtin_amdgcn_wmma_f32_16x16x32_f16(false, af[s], false, bf,
                                                 (short)0, a.v, false, false);
  }
}

__device__ __forceinline__ accf bias_acc(const float* b, int ct) {
  float bv = b[ct * 16 + (threadIdx.x & 15)];
  accf a;
#pragma unroll
  for (int v = 0; v < 8; ++v) a.f[v] = bv;
  return a;
}

__device__ __forceinline__ void copy_w(unsigned* dst, const unsigned* src, int n) {
  const uint4* s4 = (const uint4*)src;
  uint4* d4 = (uint4*)dst;
  for (int i = threadIdx.x; i < (n >> 2); i += BLK) d4[i] = s4[i];
}

// LayerNorm over 128 cols + ReLU, result packed f16 into Sh[16][64] u32.
// A[ct].f[v] holds element (m = v + 8*half, n = 16*ct + (l&15)).
__device__ __forceinline__ void ln_relu_pack(accf* A, const float* g, const float* b,
                                             unsigned* Sh) {
  int l = threadIdx.x & 31, mrow = l & 15, half = l >> 4;
#pragma unroll
  for (int v = 0; v < 8; ++v) {
    float s = 0.f, ss = 0.f;
#pragma unroll
    for (int ct = 0; ct < 8; ++ct) { float x = A[ct].f[v]; s += x; ss += x * x; }
#pragma unroll
    for (int msk = 1; msk < 16; msk <<= 1) {
      s  += __shfl_xor(s,  msk, 16);
      ss += __shfl_xor(ss, msk, 16);
    }
    float mu = s * (1.0f / 128.0f);
    float var = fmaxf(ss * (1.0f / 128.0f) - mu * mu, 0.f);
    float r = rsqrtf(var + 1e-5f);
    int m = v + half * 8;
#pragma unroll
    for (int ct = 0; ct < 8; ++ct) {
      int n = ct * 16 + mrow;
      float val = fmaxf((A[ct].f[v] - mu) * r * g[n] + b[n], 0.f);
      float o = __shfl_xor(val, 1, 32);
      if ((l & 1) == 0) Sh[m * 64 + (n >> 1)] = pack2(val, o);
    }
  }
}

// ---------------------------------------------------------------------------
// Weight pre-pack: f32 (K x 128) -> f16 pairs in WMMA-fragment order.
// out[((ct*NS + s)*32 + l)*8 + v] = pack2(W[k][n], W[k+1][n]),
//   k = 32s + 16*(l>>4) + 2v, n = 16*ct + (l&15), NS = K/32.
// ---------------------------------------------------------------------------
__global__ void pack_weights(const float* __restrict__ W, unsigned* __restrict__ out,
                             int NS) {
  int idx = blockIdx.x * blockDim.x + threadIdx.x;
  int total = 2048 * NS;
  if (idx >= total) return;
  int v = idx & 7;
  int l = (idx >> 3) & 31;
  int t = idx >> 8;            // ct*NS + s
  int s = t % NS, ct = t / NS;
  int k = 32 * s + 16 * (l >> 4) + 2 * v;
  int n = ct * 16 + (l & 15);
  out[idx] = pack2(W[k * 128 + n], W[(k + 1) * 128 + n]);
}

// ---------------------------------------------------------------------------
// Node forward: h_node = MLP(x) (stored packed f16), aggr = x @ cen_W + cen_b
// ---------------------------------------------------------------------------
__global__ __launch_bounds__(BLK) void node_fwd(
    const float* __restrict__ x,
    const float* __restrict__ nb1, const float* __restrict__ ng1,
    const float* __restrict__ nbe1, const float* __restrict__ nb2,
    const float* __restrict__ cenb,
    const unsigned* __restrict__ pw1, const unsigned* __restrict__ pw2,
    const unsigned* __restrict__ pwc,
    unsigned* __restrict__ hnode, float* __restrict__ aggr, int N_) {
  extern __shared__ unsigned smem[];
  unsigned* wbuf = smem;
  int wave = threadIdx.x >> 5, l = threadIdx.x & 31;
  unsigned* wb = smem + WBUF_U32 + wave * WAVE_U32;
  unsigned* Sx = wb + 512;
  unsigned* Sh = wb + 1536;
  int half = l >> 4, mrow = l & 15;
  int n0 = blockIdx.x * BLK_ROWS + wave * TILE_M;

  for (int j = 0; j < 16; ++j) {
    int u = j * 32 + l, m = u >> 5, c4 = u & 31;
    int node = n0 + m; if (node >= N_) node = N_ - 1;
    float4 f = ((const float4*)(x + (size_t)node * 128))[c4];
    uint2 p; p.x = pack2(f.x, f.y); p.y = pack2(f.z, f.w);
    ((uint2*)Sx)[m * 32 + c4] = p;
  }
  copy_w(wbuf, pw1, 8192);
  __syncthreads();

  v16h ax[4];
#pragma unroll
  for (int s = 0; s < 4; ++s) ax[s] = load_a(Sx, 64, s);

  accf A[8];
#pragma unroll
  for (int ct = 0; ct < 8; ++ct) {
    accf a = bias_acc(nb1, ct);
    gemm_pre(a, ax, wbuf, ct * 4, 0, 4);
    A[ct] = a;
  }
  ln_relu_pack(A, ng1, nbe1, Sh);
  __syncthreads();
  copy_w(wbuf, pw2, 8192);
  __syncthreads();

  v16h ah[4];
#pragma unroll
  for (int s = 0; s < 4; ++s) ah[s] = load_a(Sh, 64, s);

#pragma unroll
  for (int ct = 0; ct < 8; ++ct) {
    accf a = bias_acc(nb2, ct);
    gemm_pre(a, ah, wbuf, ct * 4, 0, 4);
#pragma unroll
    for (int v = 0; v < 8; ++v) {
      int m = v + half * 8, node = n0 + m;
      float val = a.f[v];
      float o = __shfl_xor(val, 1, 32);
      int n = ct * 16 + mrow;
      if (node < N_ && (l & 1) == 0)
        hnode[(size_t)node * 64 + (n >> 1)] = pack2(val, o);
    }
  }
  __syncthreads();
  copy_w(wbuf, pwc, 8192);
  __syncthreads();

#pragma unroll
  for (int ct = 0; ct < 8; ++ct) {
    accf a = bias_acc(cenb, ct);
    gemm_pre(a, ax, wbuf, ct * 4, 0, 4);   // reuse staged-x fragments
#pragma unroll
    for (int v = 0; v < 8; ++v) {
      int m = v + half * 8, node = n0 + m, n = ct * 16 + mrow;
      if (node < N_) aggr[(size_t)node * 128 + n] = a.f[v];
    }
  }
}

// ---------------------------------------------------------------------------
// Edge forward (fused): h_edge MLP, gate MLP, message GEMM, gated scatter-add
// ---------------------------------------------------------------------------
__global__ __launch_bounds__(BLK) void edge_fwd(
    const float* __restrict__ x, const int* __restrict__ eidx,
    const float* __restrict__ eattr, const float* __restrict__ ntime,
    const float* __restrict__ eb1, const float* __restrict__ eg1,
    const float* __restrict__ ebe1, const float* __restrict__ eb2,
    const float* __restrict__ gw1f, const float* __restrict__ gb1,
    const float* __restrict__ gg1, const float* __restrict__ gbe1,
    const float* __restrict__ gb2, const float* __restrict__ msgb,
    const unsigned* __restrict__ pwE1, const unsigned* __restrict__ pwE2,
    const unsigned* __restrict__ pwG1, const unsigned* __restrict__ pwG2,
    const unsigned* __restrict__ pwM,
    const unsigned* __restrict__ hnode, float* __restrict__ aggr,
    int E_, int N_) {
  extern __shared__ unsigned smem[];
  unsigned* wbuf = smem;
  int wave = threadIdx.x >> 5, l = threadIdx.x & 31;
  unsigned* wb = smem + WBUF_U32 + wave * WAVE_U32;
  unsigned* Sedge = wb;
  unsigned* Sx    = wb + 512;
  unsigned* Sh    = wb + 1536;
  int* colA  = (int*)(wb + 2560);
  int* rowA  = (int*)(wb + 2576);
  float* tA  = (float*)(wb + 2592);
  int half = l >> 4, mrow = l & 15;
  long e0 = (long)blockIdx.x * BLK_ROWS + wave * TILE_M;

  if (l < 16) {
    int e = (int)e0 + l;
    int r = eidx[e], c = eidx[E_ + e];
    rowA[l] = r; colA[l] = c; tA[l] = ntime[c];
  }
  __builtin_amdgcn_wave_barrier();

  for (int j = 0; j < 8; ++j) {
    int u = j * 32 + l, m = u >> 4, c4 = u & 15;
    float4 f = ((const float4*)(eattr + (e0 + m) * 64))[c4];
    uint2 p; p.x = pack2(f.x, f.y); p.y = pack2(f.z, f.w);
    ((uint2*)Sedge)[m * 16 + c4] = p;
  }
  for (int j = 0; j < 16; ++j) {
    int u = j * 32 + l, m = u >> 5, c4 = u & 31;
    int node = colA[m];
    float4 f = ((const float4*)(x + (size_t)node * 128))[c4];
    uint2 p; p.x = pack2(f.x, f.y); p.y = pack2(f.z, f.w);
    ((uint2*)Sx)[m * 32 + c4] = p;
  }

  // ---- phase E1: h1 = edge_attr @ W1e, LN, ReLU ----
  copy_w(wbuf, pwE1, 4096);
  __syncthreads();

  v16h aE[2], aX[4], aH[4];
#pragma unroll
  for (int s = 0; s < 2; ++s) aE[s] = load_a(Sedge, 32, s);
#pragma unroll
  for (int s = 0; s < 4; ++s) aX[s] = load_a(Sx, 64, s);

  accf A[8];
#pragma unroll
  for (int ct = 0; ct < 8; ++ct) {
    accf a = bias_acc(eb1, ct);
    gemm_pre(a, aE, wbuf, ct * 2, 0, 2);
    A[ct] = a;
  }
  ln_relu_pack(A, eg1, ebe1, Sh);
  __syncthreads();

  // ---- phase E2: h_edge = h1 @ W2e (kept in VGPRs) ----
  copy_w(wbuf, pwE2, 8192);
  __syncthreads();
#pragma unroll
  for (int s = 0; s < 4; ++s) aH[s] = load_a(Sh, 64, s);
  accf accE[8];
#pragma unroll
  for (int ct = 0; ct < 8; ++ct) {
    accf a = bias_acc(eb2, ct);
    gemm_pre(a, aH, wbuf, ct * 4, 0, 4);
    accE[ct] = a;
  }
  __syncthreads();

  // ---- phase G1: g1 = [edge_attr | x[col] | t] @ W1g, LN, ReLU ----
  copy_w(wbuf, pwG1, 12288);
  __syncthreads();
#pragma unroll
  for (int ct = 0; ct < 8; ++ct) {
    int n = ct * 16 + mrow;
    accf a = bias_acc(gb1, ct);
    gemm_pre(a, aE, wbuf, ct * 6, 0, 2);  // K 0..63   (edge_attr frags)
    gemm_pre(a, aX, wbuf, ct * 6, 2, 4);  // K 64..191 (x[col] frags)
    float wlast = gw1f[192 * 128 + n];    // K = 192 (time col, rank-1 update)
#pragma unroll
    for (int v = 0; v < 8; ++v) a.f[v] += tA[v + half * 8] * wlast;
    A[ct] = a;
  }
  ln_relu_pack(A, gg1, gbe1, Sh);
  __syncthreads();

  // ---- phase G2: gate = sigmoid(g1 @ W2g) (kept in VGPRs) ----
  copy_w(wbuf, pwG2, 8192);
  __syncthreads();
#pragma unroll
  for (int s = 0; s < 4; ++s) aH[s] = load_a(Sh, 64, s);
  accf sig[8];
#pragma unroll
  for (int ct = 0; ct < 8; ++ct) {
    accf a = bias_acc(gb2, ct);
    gemm_pre(a, aH, wbuf, ct * 4, 0, 4);
#pragma unroll
    for (int v = 0; v < 8; ++v) sig[ct].f[v] = 1.0f / (1.0f + __expf(-a.f[v]));
  }

  // ---- message input: (h_edge * h_node[col]) packed into Sh ----
  for (int j = 0; j < 8; ++j) {              // gather pre-packed h_node rows
    int u = j * 32 + l, m = u >> 4, c16 = u & 15;
    ((uint4*)Sx)[m * 16 + c16] = ((const uint4*)(hnode + (size_t)colA[m] * 64))[c16];
  }
  __builtin_amdgcn_wave_barrier();
#pragma unroll
  for (int ct = 0; ct < 8; ++ct) {
#pragma unroll
    for (int v = 0; v < 8; ++v) {
      int n = ct * 16 + mrow, m = v + half * 8;
      float hv = unpack_half(Sx[m * 64 + (n >> 1)], l & 1);
      float p = accE[ct].f[v] * hv;
      float o = __shfl_xor(p, 1, 32);
      if ((l & 1) == 0) Sh[m * 64 + (n >> 1)] = pack2(p, o);
    }
  }
  __syncthreads();

  // ---- phase MSG: msg = in @ msg_W + b, gate, scatter-add ----
  copy_w(wbuf, pwM, 8192);
  __syncthreads();
#pragma unroll
  for (int s = 0; s < 4; ++s) aH[s] = load_a(Sh, 64, s);
#pragma unroll
  for (int ct = 0; ct < 8; ++ct) {
    accf a = bias_acc(msgb, ct);
    gemm_pre(a, aH, wbuf, ct * 4, 0, 4);
#pragma unroll
    for (int v = 0; v < 8; ++v) {
      int m = v + half * 8, n = ct * 16 + mrow;
      float val = a.f[v] * sig[ct].f[v];
      unsafeAtomicAdd(&aggr[(size_t)rowA[m] * 128 + n], val);  // global_atomic_add_f32
    }
  }
}

// ---------------------------------------------------------------------------
// Output: out = ReLU(LN(aggr)) @ out_W + out_b
// ---------------------------------------------------------------------------
__global__ __launch_bounds__(BLK) void out_fwd(
    const float* __restrict__ aggr, const float* __restrict__ lng,
    const float* __restrict__ lnb, const float* __restrict__ outb,
    const unsigned* __restrict__ pwo, float* __restrict__ out, int N_) {
  extern __shared__ unsigned smem[];
  unsigned* wbuf = smem;
  int wave = threadIdx.x >> 5, l = threadIdx.x & 31;
  unsigned* Sh = smem + WBUF_U32 + wave * WAVE_U32 + 1536;
  int half = l >> 4, mrow = l & 15;
  int n0 = blockIdx.x * BLK_ROWS + wave * TILE_M;

  copy_w(wbuf, pwo, 8192);
  accf A[8];
#pragma unroll
  for (int ct = 0; ct < 8; ++ct) {
#pragma unroll
    for (int v = 0; v < 8; ++v) {
      int node = n0 + v + half * 8; if (node >= N_) node = N_ - 1;
      A[ct].f[v] = aggr[(size_t)node * 128 + ct * 16 + mrow];
    }
  }
  ln_relu_pack(A, lng, lnb, Sh);
  __syncthreads();

  v16h ah[4];
#pragma unroll
  for (int s = 0; s < 4; ++s) ah[s] = load_a(Sh, 64, s);
#pragma unroll
  for (int ct = 0; ct < 8; ++ct) {
    accf a = bias_acc(outb, ct);
    gemm_pre(a, ah, wbuf, ct * 4, 0, 4);
#pragma unroll
    for (int v = 0; v < 8; ++v) {
      int m = v + half * 8, node = n0 + m, n = ct * 16 + mrow;
      if (node < N_) out[(size_t)node * 128 + n] = a.f[v];
    }
  }
}

// ---------------------------------------------------------------------------
extern "C" void kernel_launch(void* const* d_in, const int* in_sizes, int n_in,
                              void* d_out, int out_size, void* d_ws, size_t ws_size,
                              hipStream_t stream) {
  const int N = in_sizes[IN_X] / 128;       // 50000
  const int E = in_sizes[IN_EATTR] / 64;    // 800000

  unsigned* ws = (unsigned*)d_ws;
  unsigned* hnode = ws;                            // N*64  (packed f16 pairs)
  float*    aggr  = (float*)(ws + (size_t)N * 64); // N*128 f32
  unsigned* W     = ws + (size_t)N * 192;
  unsigned* pwN1 = W;            // NS=4: 8192 u32
  unsigned* pwN2 = W + 8192;
  unsigned* pwC  = W + 16384;
  unsigned* pwE1 = W + 24576;    // NS=2: 4096 u32
  unsigned* pwE2 = W + 28672;
  unsigned* pwG1 = W + 36864;    // NS=6: 12288 u32 (first 192 rows of gate W1)
  unsigned* pwG2 = W + 49152;
  unsigned* pwM  = W + 57344;
  unsigned* pwO  = W + 65536;

  auto pack = [&](int in_idx, unsigned* dst, int NS) {
    int total = 2048 * NS;
    pack_weights<<<(total + 255) / 256, 256, 0, stream>>>(
        (const float*)d_in[in_idx], dst, NS);
  };
  pack(NN_W1, pwN1, 4);
  pack(NN_W2, pwN2, 4);
  pack(P_CENW, pwC, 4);
  pack(EN_W1, pwE1, 2);
  pack(EN_W2, pwE2, 4);
  pack(GN_W1, pwG1, 6);   // row 192 handled in-kernel as rank-1 update
  pack(GN_W2, pwG2, 4);
  pack(P_MSGW, pwM, 4);
  pack(P_OUTW, pwO, 4);

  int nblocks = (N + BLK_ROWS - 1) / BLK_ROWS;
  node_fwd<<<nblocks, BLK, SMEM_BYTES, stream>>>(
      (const float*)d_in[IN_X],
      (const float*)d_in[NN_B1], (const float*)d_in[NN_G1],
      (const float*)d_in[NN_BE1], (const float*)d_in[NN_B2],
      (const float*)d_in[P_CENB],
      pwN1, pwN2, pwC, hnode, aggr, N);

  int eblocks = (E + BLK_ROWS - 1) / BLK_ROWS;
  edge_fwd<<<eblocks, BLK, SMEM_BYTES, stream>>>(
      (const float*)d_in[IN_X], (const int*)d_in[IN_EIDX],
      (const float*)d_in[IN_EATTR], (const float*)d_in[IN_NTIME],
      (const float*)d_in[EN_B1], (const float*)d_in[EN_G1],
      (const float*)d_in[EN_BE1], (const float*)d_in[EN_B2],
      (const float*)d_in[GN_W1], (const float*)d_in[GN_B1],
      (const float*)d_in[GN_G1], (const float*)d_in[GN_BE1],
      (const float*)d_in[GN_B2], (const float*)d_in[P_MSGB],
      pwE1, pwE2, pwG1, pwG2, pwM, hnode, aggr, E, N);

  out_fwd<<<nblocks, BLK, SMEM_BYTES, stream>>>(
      aggr, (const float*)d_in[P_LNG], (const float*)d_in[P_LNB],
      (const float*)d_in[P_OUTB], pwO, (float*)d_out, N);
}
